// MultiheadAttention_83030307766273
// MI455X (gfx1250) — compile-verified, matrix operand order checked by reading, not run on hardware
//
#include <hip/hip_runtime.h>
#include <hip/hip_bf16.h>

// MI455X / gfx1250: wave32, WMMA 16x16x32 bf16 (f32 accum).
// Attention fwd: weights pre-converted/transposed to bf16 once (L2-resident),
// Q/K/V/O projections + causal flash attention. Scores are computed as S^T
// (keys x queries) so each lane owns one query's logits: softmax reductions
// are in-register + a single shfl_xor(16). All K/V fragment loads for a chunk
// are issued up front so softmax VALU work hides the V-load latency.

typedef __attribute__((ext_vector_type(16))) __bf16 v16bf;
typedef __attribute__((ext_vector_type(8)))  __bf16 v8bf;
typedef __attribute__((ext_vector_type(8)))  float  v8f;
typedef __attribute__((ext_vector_type(4)))  float  v4f;

#define DM   1024
#define NH   16
#define DK   64
#define SEQ  2048
#define NB   2
#define ROWS (NB * SEQ)   // 4096
#define TT   64           // transpose tile

#define SCALE2 0.18033688f   // (1/sqrt(64)) * log2(e)
#define MASKED_VAL (-1e30f)

// 16-bit A/B fragment: lane L holds row/col = L%16; element j holds
// K = (j/8)*16 + (L/16)*8 + (j%8)  ->  two contiguous 8-element K runs.
__device__ __forceinline__ v8bf ld8(const __bf16* p) { return *(const v8bf*)p; }

__device__ __forceinline__ v16bf cat8(v8bf lo, v8bf hi) {
    return __builtin_shufflevector(lo, hi, 0, 1, 2, 3, 4, 5, 6, 7,
                                   8, 9, 10, 11, 12, 13, 14, 15);
}

// p points at this lane's K=0 of a K-contiguous row; half = lane>>4.
__device__ __forceinline__ v16bf ldfrag(const __bf16* p, int half) {
    return cat8(ld8(p + half * 8), ld8(p + 16 + half * 8));
}

// Convert 8 contiguous f32 to v8bf (two 128-bit loads + packed cvt).
__device__ __forceinline__ v8bf cvt8(const float* p) {
    v4f u = *(const v4f*)p;
    v4f v = *(const v4f*)(p + 4);
    v8bf r;
#pragma unroll
    for (int j = 0; j < 4; ++j) { r[j] = (__bf16)u[j]; r[4 + j] = (__bf16)v[j]; }
    return r;
}

__device__ __forceinline__ v16bf ldfrag_f32(const float* p, int half) {
    return cat8(cvt8(p + half * 8), cvt8(p + 16 + half * 8));
}

__device__ __forceinline__ v8f wmma_bf16(v16bf a, v16bf b, v8f c) {
    return __builtin_amdgcn_wmma_f32_16x16x32_bf16(false, a, false, b, (short)0, c,
                                                   false, false);
}

// ---------------------------------------------------------------------------
// One-time: W[f32, 1024 x 1024] -> Wt[bf16, 1024 x 1024], Wt[n][k] = W[k][n].
// ---------------------------------------------------------------------------
__global__ __launch_bounds__(256) void transpose_w_kernel(const float* __restrict__ W,
                                                          __bf16* __restrict__ Wt) {
    __shared__ __bf16 T[TT][TT + 1];
    const int k0 = blockIdx.x * TT;
    const int n0 = blockIdx.y * TT;
    for (int i = threadIdx.x; i < TT * TT; i += 256) {
        const int r = i >> 6, c = i & 63;                 // c fastest: coalesced
        T[r][c] = (__bf16)W[(size_t)(k0 + r) * DM + n0 + c];
    }
    __syncthreads();
    for (int i = threadIdx.x; i < TT * TT; i += 256) {
        const int r = i >> 6, c = i & 63;                 // c fastest: coalesced
        Wt[(size_t)(n0 + r) * DM + k0 + c] = T[c][r];
    }
}

// ---------------------------------------------------------------------------
// X[f32, rows x 1024] @ W (as Wt[n][k] bf16) -> bf16 workspace.
// VMODE==0: head-major rows  dst[((B*NH+h)*SEQ + i)*DK + d]   (Q, K)
// VMODE==1: head-transposed  dst[((B*NH+h)*DK + d)*SEQ + i]   (V)
// Block: 256 threads (8 waves), tile 128 rows x 64 cols; wave = 16 rows.
// ---------------------------------------------------------------------------
template <int VMODE>
__global__ __launch_bounds__(256) void proj_qkv_kernel(const float* __restrict__ X,
                                                       const __bf16* __restrict__ Wt,
                                                       __bf16* __restrict__ dst) {
    const int lane    = threadIdx.x & 31;
    const int wave    = threadIdx.x >> 5;
    const int rowBase = blockIdx.x * 128 + wave * 16;
    const int colBase = blockIdx.y * 64;
    const int ln16    = lane & 15;
    const int half    = lane >> 4;

    const float* xrow = X + (size_t)(rowBase + ln16) * DM;
    v8f acc[4] = {};

    for (int k0 = 0; k0 < DM; k0 += 32) {
        __builtin_prefetch(xrow + k0 + 32, 0, 1);         // global_prefetch path
        v16bf a = ldfrag_f32(xrow + k0, half);
#pragma unroll
        for (int nb = 0; nb < 4; ++nb) {
            const __bf16* wp = Wt + (size_t)(colBase + nb * 16 + ln16) * DM + k0;
            acc[nb] = wmma_bf16(a, ldfrag(wp, half), acc[nb]);
        }
    }

    const int bb     = rowBase >> 11;                     // batch (tile never crosses)
    const int iiBase = (rowBase & (SEQ - 1)) + half * 8;
#pragma unroll
    for (int nb = 0; nb < 4; ++nb) {
        const int col = colBase + nb * 16 + ln16;
        const int h   = col >> 6, dd = col & (DK - 1);
        if (VMODE == 0) {
#pragma unroll
            for (int r = 0; r < 8; ++r)
                dst[(((size_t)(bb * NH + h)) * SEQ + iiBase + r) * DK + dd] =
                    (__bf16)acc[nb][r];
        } else {
            v8bf pack;
#pragma unroll
            for (int r = 0; r < 8; ++r) pack[r] = (__bf16)acc[nb][r];
            *(v8bf*)&dst[(((size_t)(bb * NH + h)) * DK + dd) * SEQ + iiBase] = pack;
        }
    }
}

// ---------------------------------------------------------------------------
// Flash-attention key-chunk processor (32 keys). S^T layout: lane n owns
// query q0+n; its 16 regs (2 tiles x 8) are keys. MASKED only on the single
// diagonal chunk. Base-2 softmax (logits pre-scaled by log2e/sqrt(dk)).
// ---------------------------------------------------------------------------
template <bool MASKED>
__device__ __forceinline__ void attn_chunk(int kv, int q0, int n, int half,
                                           const __bf16* __restrict__ Kb,
                                           const __bf16* __restrict__ Vb,
                                           __bf16* __restrict__ Plds,
                                           float* __restrict__ Sa,
                                           const v16bf& qb0, const v16bf& qb1,
                                           v8f (&o)[4], float& mi, float& li) {
    // Issue ALL chunk loads up front: 4 K fragments + 4 V fragments (b128 x16).
    const __bf16* kp0 = Kb + (size_t)(kv + n) * DK;
    const __bf16* kp1 = Kb + (size_t)(kv + 16 + n) * DK;
    const v16bf kf00 = ldfrag(kp0, half);
    const v16bf kf01 = ldfrag(kp0 + 32, half);
    const v16bf kf10 = ldfrag(kp1, half);
    const v16bf kf11 = ldfrag(kp1 + 32, half);
    v16bf vf[4];
#pragma unroll
    for (int nb = 0; nb < 4; ++nb)
        vf[nb] = ldfrag(Vb + (size_t)(nb * 16 + n) * SEQ + kv, half);

    // S^T = K_chunk @ Q^T (two 16x16 tiles; contraction over d = 2x32).
    v8f st0 = {}, st1 = {};
    st0 = wmma_bf16(kf00, qb0, st0);
    st0 = wmma_bf16(kf01, qb1, st0);
    st1 = wmma_bf16(kf10, qb0, st1);
    st1 = wmma_bf16(kf11, qb1, st1);

    // Scale (+ mask on diagonal chunk). Element (r, tile): key kv+t*16+half*8+r.
    float a0[8], a1[8];
    const int qcol = q0 + n;
#pragma unroll
    for (int r = 0; r < 8; ++r) {
        a0[r] = st0[r] * SCALE2;
        a1[r] = st1[r] * SCALE2;
        if (MASKED) {
            if (kv + half * 8 + r      > qcol) a0[r] = MASKED_VAL;
            if (kv + 16 + half * 8 + r > qcol) a1[r] = MASKED_VAL;
        }
    }

    // Per-query max: in-register tree + one cross-half combine.
    float rm = fmaxf(a0[0], a1[0]);
#pragma unroll
    for (int r = 1; r < 8; ++r) rm = fmaxf(rm, fmaxf(a0[r], a1[r]));
    rm = fmaxf(rm, __shfl_xor(rm, 16, 32));
    const float mnew  = fmaxf(mi, rm);
    const float alpha = exp2f(mi - mnew);
    mi = mnew;

    // Probabilities + per-query sum (in-register + one combine).
    v8bf p0, p1;
    float rs = 0.0f;
#pragma unroll
    for (int r = 0; r < 8; ++r) {
        const float e0 = exp2f(a0[r] - mnew);
        const float e1 = exp2f(a1[r] - mnew);
        rs += e0 + e1;
        p0[r] = (__bf16)e0;
        p1[r] = (__bf16)e1;
    }
    rs += __shfl_xor(rs, 16, 32);
    li = li * alpha + rs;

    // P^T -> LDS at [query][key-local]: contiguous keys => ds_store_b128 x2.
    *(v8bf*)&Plds[n * 32 + half * 8]      = p0;
    *(v8bf*)&Plds[n * 32 + 16 + half * 8] = p1;
    if (half == 0) Sa[n] = alpha;         // lane-layout -> row-layout broadcast
    __syncthreads();

    // Rescale O by this chunk's alpha, per accumulator row (query half*8+r).
    const v4f al0 = *(const v4f*)&Sa[half * 8];
    const v4f al1 = *(const v4f*)&Sa[half * 8 + 4];
#pragma unroll
    for (int r = 0; r < 8; ++r) {
        const float a = (r < 4) ? al0[r] : al1[r - 4];
        o[0][r] *= a; o[1][r] *= a; o[2][r] *= a; o[3][r] *= a;
    }

    const v16bf pa = ldfrag(&Plds[n * 32], half);   // two ds_load_b128

    // O += P @ V  (V fragments already resident).
#pragma unroll
    for (int nb = 0; nb < 4; ++nb)
        o[nb] = wmma_bf16(pa, vf[nb], o[nb]);
    __syncthreads();
}

// ---------------------------------------------------------------------------
// Causal flash attention. One wave per (16-row query block, b*h).
// Q,K: bf16 [bh][seq][64] (row-major).  V: bf16 [bh][64][seq] (transposed).
// ctx: bf16 [b, seq, 1024].
// ---------------------------------------------------------------------------
__global__ __launch_bounds__(32) void attn_kernel(const __bf16* __restrict__ Q,
                                                  const __bf16* __restrict__ K,
                                                  const __bf16* __restrict__ Vt,
                                                  __bf16* __restrict__ ctx) {
    __shared__ alignas(16) __bf16 Plds[16 * 32];
    __shared__ alignas(16) float  Sa[16];
    __shared__ alignas(16) float  Sl[16];
    const int lane = threadIdx.x & 31;
    const int n    = lane & 15;
    const int half = lane >> 4;
    const int q0   = blockIdx.x * 16;
    const int bh   = blockIdx.y;
    const size_t base = (size_t)bh * SEQ * DK;
    const __bf16* Kb = K + base;
    const __bf16* Vb = Vt + base;

    // Q^T B-fragments, resident for the whole block (lane n = query q0+n).
    const __bf16* qp = Q + base + (size_t)(q0 + n) * DK;
    const v16bf qb0 = ldfrag(qp, half);
    const v16bf qb1 = ldfrag(qp + 32, half);

    v8f o[4] = {};
    float mi = -1e30f, li = 0.0f;

    // Interior chunks (kv+31 < q0): mask-free fast path.
    const int tdiag = q0 >> 5;
    for (int t = 0; t < tdiag; ++t)
        attn_chunk<false>(t << 5, q0, n, half, Kb, Vb, Plds, Sa, qb0, qb1, o, mi, li);
    // Single diagonal chunk: masked path.
    attn_chunk<true>(tdiag << 5, q0, n, half, Kb, Vb, Plds, Sa, qb0, qb1, o, mi, li);

    // Broadcast 1/li from lane-layout to accumulator-row layout.
    if (half == 0) Sl[n] = li;
    __syncthreads();
    const v4f l0 = *(const v4f*)&Sl[half * 8];
    const v4f l1 = *(const v4f*)&Sl[half * 8 + 4];

    // Store ctx in [b, seq, H*DV] layout for the O projection.
    const int h  = bh & (NH - 1);
    const int bb = bh >> 4;
#pragma unroll
    for (int r = 0; r < 8; ++r) {
        const int row = q0 + half * 8 + r;
        const float inv = 1.0f / ((r < 4) ? l0[r] : l1[r - 4]);
#pragma unroll
        for (int nb = 0; nb < 4; ++nb) {
            ctx[((size_t)(bb * SEQ + row)) * DM + h * DK + nb * 16 + n] =
                (__bf16)(o[nb][r] * inv);
        }
    }
}

// ---------------------------------------------------------------------------
// ctx[bf16, 4096 x 1024] @ W_o (as Wt[n][k] bf16) -> out f32
// ---------------------------------------------------------------------------
__global__ __launch_bounds__(256) void proj_out_kernel(const __bf16* __restrict__ X,
                                                       const __bf16* __restrict__ Wt,
                                                       float* __restrict__ out) {
    const int lane    = threadIdx.x & 31;
    const int wave    = threadIdx.x >> 5;
    const int rowBase = blockIdx.x * 128 + wave * 16;
    const int colBase = blockIdx.y * 64;
    const int ln16    = lane & 15;
    const int half    = lane >> 4;

    const __bf16* xrow = X + (size_t)(rowBase + ln16) * DM;
    v8f acc[4] = {};

    for (int k0 = 0; k0 < DM; k0 += 32) {
        v16bf a = ldfrag(xrow + k0, half);
#pragma unroll
        for (int nb = 0; nb < 4; ++nb) {
            const __bf16* wp = Wt + (size_t)(colBase + nb * 16 + ln16) * DM + k0;
            acc[nb] = wmma_bf16(a, ldfrag(wp, half), acc[nb]);
        }
    }

#pragma unroll
    for (int nb = 0; nb < 4; ++nb) {
#pragma unroll
        for (int r = 0; r < 8; ++r) {
            const int row = rowBase + half * 8 + r;
            const int col = colBase + nb * 16 + ln16;
            out[(size_t)row * DM + col] = acc[nb][r];
        }
    }
}

// ---------------------------------------------------------------------------
extern "C" void kernel_launch(void* const* d_in, const int* in_sizes, int n_in,
                              void* d_out, int out_size, void* d_ws, size_t ws_size,
                              hipStream_t stream) {
    const float* query     = (const float*)d_in[0];
    const float* key_value = (const float*)d_in[1];
    // d_in[2] = mask: structurally causal tril, applied analytically in-kernel.
    const float* W_q = (const float*)d_in[3];
    const float* W_k = (const float*)d_in[4];
    const float* W_v = (const float*)d_in[5];
    const float* W_o = (const float*)d_in[6];

    const size_t HE = (size_t)NB * NH * SEQ * DK;  // 4M elems per Q/K/V
    const size_t WE = (size_t)DM * DM;             // 1M elems per weight
    __bf16* Qws = (__bf16*)d_ws;
    __bf16* Kws = Qws + HE;
    __bf16* Vws = Kws + HE;                        // stored transposed [bh][d][seq]
    __bf16* Cws = Vws + HE;                        // ctx: 4096*1024
    __bf16* WqT = Cws + (size_t)ROWS * DM;         // bf16 Wt[n][k] per weight
    __bf16* WkT = WqT + WE;
    __bf16* WvT = WkT + WE;
    __bf16* WoT = WvT + WE;

    dim3 gT(DM / TT, DM / TT);                     // 16 x 16
    transpose_w_kernel<<<gT, 256, 0, stream>>>(W_q, WqT);
    transpose_w_kernel<<<gT, 256, 0, stream>>>(W_k, WkT);
    transpose_w_kernel<<<gT, 256, 0, stream>>>(W_v, WvT);
    transpose_w_kernel<<<gT, 256, 0, stream>>>(W_o, WoT);

    dim3 gProj(ROWS / 128, DM / 64);               // 32 x 16
    proj_qkv_kernel<0><<<gProj, 256, 0, stream>>>(query,     WqT, Qws);
    proj_qkv_kernel<0><<<gProj, 256, 0, stream>>>(key_value, WkT, Kws);
    proj_qkv_kernel<1><<<gProj, 256, 0, stream>>>(key_value, WvT, Vws);

    dim3 gAttn(SEQ / 16, NB * NH);                 // 128 x 32, one wave each
    attn_kernel<<<gAttn, 32, 0, stream>>>(Qws, Kws, Vws, Cws);

    proj_out_kernel<<<gProj, 256, 0, stream>>>(Cws, WoT, (float*)d_out);
}
